// TransformerEncoderLayer_4810363372627
// MI455X (gfx1250) — matrix operand order repaired and checked by reading, not compiled, not run on hardware
//
#include <hip/hip_runtime.h>
#include <math.h>

// ---------------------------------------------------------------------------
// MI455X (gfx1250) implementation. wave32.
//   * all GEMMs: v_wmma_f32_16x16x32_bf16, 64x64 block tile, K-step 64
//   * GEMM tile staging: Tensor Data Mover (tensor_load_to_lds) + s_wait_tensorcnt
//   * canonical GEMM form C = A[MxK] @ Bt[NxK]^T  (every B pre-transposed once)
// ---------------------------------------------------------------------------

typedef __attribute__((ext_vector_type(16))) __bf16    v16bf;
typedef __attribute__((ext_vector_type(8)))  float     v8f;
typedef __attribute__((ext_vector_type(4)))  unsigned  u32x4;
typedef __attribute__((ext_vector_type(8)))  int       i32x8;
typedef __attribute__((ext_vector_type(4)))  int       i32x4;

#define N_ATOMS 512
#define E_EDGES 4096
#define H_DIM   128
#define HH_DIM  1024
#define RBFK    64

// ---------------------------- elementwise kernels --------------------------

__global__ void k_f32_to_bf16(const float* __restrict__ in, __bf16* __restrict__ out, int n) {
  int i = blockIdx.x * blockDim.x + threadIdx.x;
  if (i < n) out[i] = (__bf16)in[i];
}

// out[n*Kd + k] = in[k*Nd + n]  (f32 [Kd x Nd] -> bf16 [Nd x Kd])
__global__ void k_f32_to_bf16_T(const float* __restrict__ in, __bf16* __restrict__ out,
                                int Kd, int Nd) {
  int idx = blockIdx.x * blockDim.x + threadIdx.x;
  if (idx >= Kd * Nd) return;
  int k = idx / Nd, n = idx % Nd;
  out[(size_t)n * Kd + k] = (__bf16)in[idx];
}

__global__ void k_zero_f32(float* __restrict__ p, int n) {
  int i = blockIdx.x * blockDim.x + threadIdx.x;
  if (i < n) p[i] = 0.f;
}

// ai = atom_embs[tgt] + ew ; aj = atom_embs[src] + ew   (bf16, E x H)
__global__ void k_edge_embed(const float* __restrict__ atom_embs,
                             const float* __restrict__ edge_weight,
                             const int* __restrict__ ei,
                             __bf16* __restrict__ ai, __bf16* __restrict__ aj) {
  int idx = blockIdx.x * blockDim.x + threadIdx.x;
  if (idx >= E_EDGES * H_DIM) return;
  int e = idx >> 7, h = idx & (H_DIM - 1);
  int s = ei[e], t = ei[E_EDGES + e];
  float ew = edge_weight[e];
  ai[idx] = (__bf16)(atom_embs[t * H_DIM + h] + ew);
  aj[idx] = (__bf16)(atom_embs[s * H_DIM + h] + ew);
}

// rbf features (E x 64, bf16)
__global__ void k_rbf(const float* __restrict__ pos, const int* __restrict__ ei,
                      __bf16* __restrict__ rbf) {
  int idx = blockIdx.x * blockDim.x + threadIdx.x;
  if (idx >= E_EDGES * RBFK) return;
  int e = idx >> 6, k = idx & (RBFK - 1);
  int s = ei[e], t = ei[E_EDGES + e];
  float dx = pos[t * 3 + 0] - pos[s * 3 + 0];
  float dy = pos[t * 3 + 1] - pos[s * 3 + 1];
  float dz = pos[t * 3 + 2] - pos[s * 3 + 2];
  float d = sqrtf(dx * dx + dy * dy + dz * dz);
  float x = d * 0.1f;
  float x3 = x * x * x, x4 = x3 * x, x5 = x4 * x;
  float cut = (x < 1.f) ? (1.f - 6.f * x5 + 15.f * x4 - 10.f * x3) : 0.f;
  float emc = __expf(-10.f);
  float c = 1.f + (emc - 1.f) * ((float)k / 63.f);
  float w = 0.5f / ((1.f - emc) / 64.f);
  w = w * w;
  float dd = __expf(-d) - c;
  rbf[idx] = (__bf16)(cut * __expf(-w * dd * dd));
}

// edge_ij = concat(hi+hj, hi-hj, hi*hj)  (E x 384, bf16)
__global__ void k_edge_ij(const float* __restrict__ hi, const float* __restrict__ hj,
                          __bf16* __restrict__ eij) {
  int idx = blockIdx.x * blockDim.x + threadIdx.x;
  if (idx >= E_EDGES * H_DIM) return;
  int e = idx >> 7, h = idx & (H_DIM - 1);
  float a = hi[idx], b = hj[idx];
  size_t base = (size_t)e * (3 * H_DIM) + h;
  eij[base]             = (__bf16)(a + b);
  eij[base + H_DIM]     = (__bf16)(a - b);
  eij[base + 2 * H_DIM] = (__bf16)(a * b);
}

// Segmented softmax over columns (segments = src[j]) for one row i per block.
__global__ void k_seg_softmax(const float* __restrict__ logits,
                              const int* __restrict__ src,
                              __bf16* __restrict__ prod) {
  __shared__ float    eL[E_EDGES];
  __shared__ unsigned smax[N_ATOMS];
  __shared__ float    ssum[N_ATOMS];
  int i = blockIdx.x, t = threadIdx.x;
  for (int s = t; s < N_ATOMS; s += blockDim.x) { smax[s] = 0u; ssum[s] = 0.f; }
  __syncthreads();
  const float* row = logits + (size_t)i * E_EDGES;
  for (int j = t; j < E_EDGES; j += blockDim.x) {
    int fi = __float_as_int(row[j]);
    unsigned u = (fi >= 0) ? ((unsigned)fi | 0x80000000u) : ~(unsigned)fi;
    atomicMax(&smax[src[j]], u);
  }
  __syncthreads();
  for (int j = t; j < E_EDGES; j += blockDim.x) {
    unsigned u = smax[src[j]];
    float m = __int_as_float((u & 0x80000000u) ? (int)(u ^ 0x80000000u) : (int)~u);
    float e = __expf(row[j] - m);
    eL[j] = e;
    atomicAdd(&ssum[src[j]], e);
  }
  __syncthreads();
  __bf16* out = prod + (size_t)i * E_EDGES;
  for (int j = t; j < E_EDGES; j += blockDim.x)
    out[j] = (__bf16)(eL[j] / ssum[src[j]]);
}

// agg[tgt[e]] += msg[e]
__global__ void k_scatter_add(const float* __restrict__ msg, const int* __restrict__ tgt,
                              float* __restrict__ agg) {
  int idx = blockIdx.x * blockDim.x + threadIdx.x;
  if (idx >= E_EDGES * H_DIM) return;
  int e = idx >> 7, h = idx & (H_DIM - 1);
  atomicAdd(&agg[tgt[e] * H_DIM + h], msg[idx]);
}

// per-row LayerNorm over H=128; optional fp32 and bf16 outputs
__global__ void k_layernorm(const float* __restrict__ x, const float* __restrict__ g,
                            const float* __restrict__ b,
                            float* __restrict__ outF, __bf16* __restrict__ outB) {
  __shared__ float red[H_DIM];
  int r = blockIdx.x, t = threadIdx.x;
  float v = x[r * H_DIM + t];
  red[t] = v;
  __syncthreads();
  for (int s = H_DIM / 2; s > 0; s >>= 1) { if (t < s) red[t] += red[t + s]; __syncthreads(); }
  float mu = red[0] * (1.f / H_DIM);
  __syncthreads();
  float dv = v - mu;
  red[t] = dv * dv;
  __syncthreads();
  for (int s = H_DIM / 2; s > 0; s >>= 1) { if (t < s) red[t] += red[t + s]; __syncthreads(); }
  float var = red[0] * (1.f / H_DIM);
  float y = dv * rsqrtf(var + 1e-5f) * g[t] + b[t];
  if (outF) outF[r * H_DIM + t] = y;
  if (outB) outB[r * H_DIM + t] = (__bf16)y;
}

__global__ void k_softplus(const float* __restrict__ in, float* __restrict__ outF,
                           __bf16* __restrict__ outB, int n) {
  int i = blockIdx.x * blockDim.x + threadIdx.x;
  if (i >= n) return;
  float x = in[i];
  float y = (x > 20.f) ? x : log1pf(__expf(x));
  if (outF) outF[i] = y;
  if (outB) outB[i] = (__bf16)y;
}

// ------------------------- TDM tile load (64 x 64 bf16) ---------------------
// Loads a 64x64 bf16 tile (row stride = rowStride elements) from global into
// LDS at byte offset lds_off, with LDS row stride 72 halves (pad: +4 dwords
// after every 32 dwords).  D# layout per CDNA5 ISA §8.3/8.4.
__device__ __forceinline__ void tdm_load_tile(const __bf16* gtile, unsigned lds_off,
                                              int rowsTotal, int colsTotal) {
  unsigned long long ga = (unsigned long long)(uintptr_t)gtile;
  u32x4 g0;
  g0[0] = 1u;                                            // count=1, user mode
  g0[1] = lds_off;                                       // lds_addr (bytes)
  g0[2] = (unsigned)(ga & 0xFFFFFFFFu);                  // global_addr[31:0]
  g0[3] = (unsigned)((ga >> 32) & 0x01FFFFFFu)           // global_addr[56:32]
        | 0x80000000u;                                   // type=2 (image)
  i32x8 g1;
  g1[0] = (1 << 16)                                      // data_size = 2 bytes
        | (1 << 20)                                      // pad_enable
        | (4 << 22)                                      // pad_interval: 32 dwords
        | (3 << 25);                                     // pad_amount: 4 dwords
  g1[1] = (int)(((unsigned)colsTotal & 0xFFFFu) << 16);  // tensor_dim0[15:0]
  g1[2] = (int)((((unsigned)colsTotal >> 16) & 0xFFFFu)  // tensor_dim0[31:16]
        | (((unsigned)rowsTotal & 0xFFFFu) << 16));      // tensor_dim1[15:0]
  g1[3] = (int)((((unsigned)rowsTotal >> 16) & 0xFFFFu)  // tensor_dim1[31:16]
        | (64u << 16));                                  // tile_dim0 = 64
  g1[4] = 64;                                            // tile_dim1 = 64
  g1[5] = colsTotal;                                     // tensor_dim0_stride[31:0]
  g1[6] = 0;                                             // stride hi / dim1_stride
  g1[7] = 0;
  i32x4 z4 = {0, 0, 0, 0};
#if defined(__clang_major__) && (__clang_major__ >= 23)
  i32x8 z8 = {0, 0, 0, 0, 0, 0, 0, 0};
  __builtin_amdgcn_tensor_load_to_lds(g0, g1, z4, z4, z8, 0);
#else
  __builtin_amdgcn_tensor_load_to_lds(g0, g1, z4, z4, 0);
#endif
}

// ------------------------------- WMMA GEMM ----------------------------------
// C[M,N] = alpha * A[MxK] @ Bt[NxK]^T (+bias[N]) (+C if accumulate)
// M,N multiples of 64; K multiple of 64. Block = 128 threads (4 waves).
__global__ __launch_bounds__(128)
void gemm_wmma_bf16(const __bf16* __restrict__ A, const __bf16* __restrict__ Bt,
                    const float* __restrict__ bias,
                    float* __restrict__ Cf, __bf16* __restrict__ Cb,
                    int M, int N, int K, int accumulate, float alpha) {
  __shared__ __align__(16) __bf16 lA[64][72];   // 72-half row stride (TDM pad)
  __shared__ __align__(16) __bf16 lB[64][72];

  const int tid  = threadIdx.x;
  const int lane = tid & 31;
  const int wv   = tid >> 5;      // 0..3
  const int hl   = lane >> 4;     // 0/1
  const int l16  = lane & 15;
  const int m0 = blockIdx.y * 64;
  const int n0 = blockIdx.x * 64;

  const unsigned ldsA = (unsigned)(size_t)&lA[0][0];
  const unsigned ldsB = (unsigned)(size_t)&lB[0][0];

  v8f acc[4];
#pragma unroll
  for (int nt = 0; nt < 4; ++nt)
#pragma unroll
    for (int r = 0; r < 8; ++r) acc[nt][r] = 0.f;

  union Frag { v16bf v; unsigned u[8]; };

  for (int k0 = 0; k0 < K; k0 += 64) {
    if (wv == 0) {  // one wave drives the DMA; TENSORcnt is per-wave
      tdm_load_tile(A  + (size_t)m0 * K + k0, ldsA, M, K);
      tdm_load_tile(Bt + (size_t)n0 * K + k0, ldsB, N, K);
      __builtin_amdgcn_s_wait_tensorcnt(0);
    }
    __syncthreads();

    // A frag: 16-bit A 16x32 layout — K(v,half,p) = (v>>2)*16 + half*8 + (v&3)*2 + p
    Frag a0, a1;
    const int mrow = wv * 16 + l16;
#pragma unroll
    for (int q = 0; q < 8; ++q) {
      int kk = ((q >> 2) << 4) + (hl << 3) + ((q & 3) << 1);
      a0.u[q] = *(const unsigned*)&lA[mrow][kk];
      a1.u[q] = *(const unsigned*)&lA[mrow][32 + kk];
    }
#pragma unroll
    for (int nt = 0; nt < 4; ++nt) {
      // B frag: 16-bit B 32x16 layout — K(v,half,p) = (v>>2)*16 + half*8 + (v&3)*2 + p, N = lane%16
      Frag b0, b1;
      const int col = nt * 16 + l16;
#pragma unroll
      for (int q = 0; q < 8; ++q) {
        int kk = ((q >> 2) << 4) + (hl << 3) + ((q & 3) << 1);
        b0.u[q] = *(const unsigned*)&lB[col][kk];
        b1.u[q] = *(const unsigned*)&lB[col][32 + kk];
      }
      acc[nt] = __builtin_amdgcn_wmma_f32_16x16x32_bf16(
          false, a0.v, false, b0.v, (short)0, acc[nt], false, false);
      acc[nt] = __builtin_amdgcn_wmma_f32_16x16x32_bf16(
          false, a1.v, false, b1.v, (short)0, acc[nt], false, false);
    }
    __syncthreads();
  }

  // epilogue: C layout — lane(0-15)=N, VGPR r => M = r + 8*(lane>>4)
#pragma unroll
  for (int nt = 0; nt < 4; ++nt) {
#pragma unroll
    for (int r = 0; r < 8; ++r) {
      int gm = m0 + wv * 16 + r + 8 * hl;
      int gn = n0 + nt * 16 + l16;
      float v = alpha * acc[nt][r];
      if (bias) v += bias[gn];
      size_t o = (size_t)gm * N + gn;
      if (accumulate) v += Cf[o];
      if (Cb) Cb[o] = (__bf16)v;
      else    Cf[o] = v;
    }
  }
}

// ------------------------------- launch -------------------------------------

extern "C" void kernel_launch(void* const* d_in, const int* in_sizes, int n_in,
                              void* d_out, int out_size, void* d_ws, size_t ws_size,
                              hipStream_t stream) {
  const float* atom_embs = (const float*)d_in[0];
  const float* pos       = (const float*)d_in[1];
  const float* edge_w    = (const float*)d_in[2];
  const int*   ei        = (const int*)d_in[3];     // [2,E]: row0=src, row1=tgt
  const float* Wq   = (const float*)d_in[4];
  const float* Wk   = (const float*)d_in[5];
  const float* Wv   = (const float*)d_in[6];
  const float* li_w = (const float*)d_in[7];
  const float* li_b = (const float*)d_in[8];
  const float* lj_w = (const float*)d_in[9];
  const float* lj_b = (const float*)d_in[10];
  const float* el_w = (const float*)d_in[11];
  const float* el_b = (const float*)d_in[12];
  const float* rl_w = (const float*)d_in[13];
  const float* rl_b = (const float*)d_in[14];
  const float* ol_w = (const float*)d_in[15];
  const float* ol_b = (const float*)d_in[16];
  const float* ln_g = (const float*)d_in[17];
  const float* ln_b = (const float*)d_in[18];
  const float* f1_w = (const float*)d_in[19];
  const float* f1_b = (const float*)d_in[20];
  const float* f2_w = (const float*)d_in[21];
  const float* f2_b = (const float*)d_in[22];
  const float* f3_w = (const float*)d_in[23];
  const float* f3_b = (const float*)d_in[24];
  float* out = (float*)d_out;

  char* base = (char*)d_ws;
  size_t off = 0;
  auto alloc = [&](size_t bytes) -> void* {
    void* p = base + off;
    off += (bytes + 255) & ~(size_t)255;
    return p;
  };
  const size_t EH = (size_t)E_EDGES * H_DIM, EHH = (size_t)E_EDGES * HH_DIM;
  const size_t EE = (size_t)E_EDGES * E_EDGES, NH = (size_t)N_ATOMS * H_DIM;

  __bf16* ai_bf   = (__bf16*)alloc(EH * 2);
  __bf16* aj_bf   = (__bf16*)alloc(EH * 2);
  __bf16* rbf_bf  = (__bf16*)alloc((size_t)E_EDGES * RBFK * 2);
  // pre-transposed bf16 weights ([N x K] for C = A @ W)
  __bf16* WqT  = (__bf16*)alloc((size_t)HH_DIM * H_DIM * 2);
  __bf16* WkT  = (__bf16*)alloc((size_t)HH_DIM * H_DIM * 2);
  __bf16* WvT  = (__bf16*)alloc((size_t)HH_DIM * H_DIM * 2);
  __bf16* liT  = (__bf16*)alloc((size_t)H_DIM * H_DIM * 2);
  __bf16* ljT  = (__bf16*)alloc((size_t)H_DIM * H_DIM * 2);
  __bf16* elT  = (__bf16*)alloc((size_t)HH_DIM * 3 * H_DIM * 2);
  __bf16* rlT  = (__bf16*)alloc((size_t)HH_DIM * RBFK * 2);
  __bf16* olT  = (__bf16*)alloc((size_t)H_DIM * HH_DIM * 2);
  __bf16* f1T  = (__bf16*)alloc((size_t)H_DIM * H_DIM * 2);
  __bf16* f2T  = (__bf16*)alloc((size_t)H_DIM * H_DIM * 2);
  __bf16* f3T  = (__bf16*)alloc((size_t)H_DIM * H_DIM * 2);
  __bf16* Qbf     = (__bf16*)alloc(EHH * 2);
  __bf16* Kbf     = (__bf16*)alloc(EHH * 2);      // also serves as Bt for logits
  float*  feats   = (float*) alloc(EHH * 4);
  __bf16* featsbf = (__bf16*)alloc(EHH * 2);
  float*  hi      = (float*) alloc(EH * 4);
  float*  hj      = (float*) alloc(EH * 4);
  __bf16* eij_bf  = (__bf16*)alloc(EH * 3 * 2);
  float*  logits  = (float*) alloc(EE * 4);
  __bf16* prod_bf = (__bf16*)alloc(EE * 2);
  float*  fo      = (float*) alloc(EH * 4);
  __bf16* foT_bf  = (__bf16*)alloc(EH * 2);       // [H x E]
  float*  msg     = (float*) alloc(EH * 4);
  float*  agg     = (float*) alloc(NH * 4);
  float*  h       = (float*) alloc(NH * 4);
  __bf16* h_bf    = (__bf16*)alloc(NH * 2);
  float*  t1      = (float*) alloc(NH * 4);
  __bf16* t1b     = (__bf16*)alloc(NH * 2);
  float*  t2      = (float*) alloc(NH * 4);
  __bf16* t2b     = (__bf16*)alloc(NH * 2);
  float*  t3      = (float*) alloc(NH * 4);
  (void)ws_size; (void)n_in; (void)in_sizes; (void)out_size;

  auto cdiv = [](long long a, long long b) { return (unsigned)((a + b - 1) / b); };
#define CONV(src_, dst_, n_) \
  k_f32_to_bf16<<<cdiv((long long)(n_), 256), 256, 0, stream>>>((src_), (dst_), (int)(n_))
#define CONVT(src_, dst_, Kd_, Nd_) \
  k_f32_to_bf16_T<<<cdiv((long long)(Kd_) * (Nd_), 256), 256, 0, stream>>>((src_), (dst_), (Kd_), (Nd_))
#define GEMM(A_, Bt_, bias_, Cf_, Cb_, M_, N_, K_, acc_, al_) \
  gemm_wmma_bf16<<<dim3(cdiv((N_), 64), cdiv((M_), 64)), 128, 0, stream>>>( \
      (A_), (Bt_), (bias_), (Cf_), (Cb_), (M_), (N_), (K_), (acc_), (al_))

  // weights -> transposed bf16  (W [K x N] -> Wt [N x K])
  CONVT(Wq,   WqT, H_DIM, HH_DIM);
  CONVT(Wk,   WkT, H_DIM, HH_DIM);
  CONVT(Wv,   WvT, H_DIM, HH_DIM);
  CONVT(li_w, liT, H_DIM, H_DIM);
  CONVT(lj_w, ljT, H_DIM, H_DIM);
  CONVT(el_w, elT, 3 * H_DIM, HH_DIM);
  CONVT(rl_w, rlT, RBFK, HH_DIM);
  CONVT(ol_w, olT, HH_DIM, H_DIM);
  CONVT(f1_w, f1T, H_DIM, H_DIM);
  CONVT(f2_w, f2T, H_DIM, H_DIM);
  CONVT(f3_w, f3T, H_DIM, H_DIM);

  // edge features
  k_edge_embed<<<cdiv(EH, 256), 256, 0, stream>>>(atom_embs, edge_w, ei, ai_bf, aj_bf);
  k_rbf<<<cdiv((size_t)E_EDGES * RBFK, 256), 256, 0, stream>>>(pos, ei, rbf_bf);

  // projections
  GEMM(ai_bf, WqT, nullptr, nullptr, Qbf, E_EDGES, HH_DIM, H_DIM, 0, 1.f);
  GEMM(ai_bf, WkT, nullptr, nullptr, Kbf, E_EDGES, HH_DIM, H_DIM, 0, 1.f);
  GEMM(ai_bf, WvT, nullptr, feats,  nullptr, E_EDGES, HH_DIM, H_DIM, 0, 1.f);  // feats = V
  GEMM(ai_bf, liT, li_b, hi, nullptr, E_EDGES, H_DIM, H_DIM, 0, 1.f);
  GEMM(aj_bf, ljT, lj_b, hj, nullptr, E_EDGES, H_DIM, H_DIM, 0, 1.f);

  k_edge_ij<<<cdiv(EH, 256), 256, 0, stream>>>(hi, hj, eij_bf);

  // feats += edge_ij@el_w + el_b ; feats += rbf@rl_w + rl_b
  GEMM(eij_bf, elT, el_b, feats, nullptr, E_EDGES, HH_DIM, 3 * H_DIM, 1, 1.f);
  GEMM(rbf_bf, rlT, rl_b, feats, nullptr, E_EDGES, HH_DIM, RBFK,      1, 1.f);
  CONV(feats, featsbf, EHH);

  // fo = feats @ ol_w + ol_b ; then transpose for the prod@fo GEMM
  GEMM(featsbf, olT, ol_b, fo, nullptr, E_EDGES, H_DIM, HH_DIM, 0, 1.f);
  CONVT(fo, foT_bf, E_EDGES, H_DIM);

  // logits = (Q @ K^T) * H^-0.5   (Kbf is already [N=E x K=HH] = Bt)
  const float scale = 1.0f / sqrtf((float)H_DIM);
  GEMM(Qbf, Kbf, nullptr, logits, nullptr, E_EDGES, E_EDGES, HH_DIM, 0, scale);

  // segmented softmax over columns grouped by src
  k_seg_softmax<<<E_EDGES, 256, 0, stream>>>(logits, ei, prod_bf);

  // msg = prod @ fo   (Bt = fo^T, [H x E])
  GEMM(prod_bf, foT_bf, nullptr, msg, nullptr, E_EDGES, H_DIM, E_EDGES, 0, 1.f);

  // agg = segment_sum(msg, tgt)
  k_zero_f32<<<cdiv(NH, 256), 256, 0, stream>>>(agg, (int)NH);
  k_scatter_add<<<cdiv(EH, 256), 256, 0, stream>>>(msg, ei + E_EDGES, agg);

  // LN + softplus MLP + LN
  k_layernorm<<<N_ATOMS, H_DIM, 0, stream>>>(agg, ln_g, ln_b, h, h_bf);
  GEMM(h_bf, f1T, f1_b, t1, nullptr, N_ATOMS, H_DIM, H_DIM, 0, 1.f);
  k_softplus<<<cdiv(NH, 256), 256, 0, stream>>>(t1, nullptr, t1b, (int)NH);
  GEMM(t1b, f2T, f2_b, t2, nullptr, N_ATOMS, H_DIM, H_DIM, 0, 1.f);
  k_softplus<<<cdiv(NH, 256), 256, 0, stream>>>(t2, nullptr, t2b, (int)NH);
  GEMM(t2b, f3T, f3_b, t3, nullptr, N_ATOMS, H_DIM, H_DIM, 0, 1.f);
  k_softplus<<<cdiv(NH, 256), 256, 0, stream>>>(t3, t3, nullptr, (int)NH);
  k_layernorm<<<N_ATOMS, H_DIM, 0, stream>>>(t3, ln_g, ln_b, out, nullptr);

#undef CONV
#undef CONVT
#undef GEMM
}
